// GatNet_42322607735204
// MI455X (gfx1250) — matrix operand order repaired and checked by reading, not compiled, not run on hardware
//
#include <hip/hip_runtime.h>
#include <hip/hip_bf16.h>
#include <math.h>

#define Nn   8192
#define FIN  512
#define Hh   8
#define C1n  8
#define HC1  64          // Hh*C1n
#define NCn  16
#define Ee   262144
#define ETot (Ee + Nn)   // edges + self loops
#define NEG_SLOPE 0.2f

typedef __attribute__((ext_vector_type(8)))  __bf16 v8bf;
typedef __attribute__((ext_vector_type(16))) __bf16 v16bf;
typedef __attribute__((ext_vector_type(8)))  float  v8f;

__device__ __forceinline__ unsigned short f2bf(float f) {
  unsigned u = __float_as_uint(f);
  unsigned r = 0x7FFFu + ((u >> 16) & 1u);   // round to nearest even
  return (unsigned short)((u + r) >> 16);
}

__device__ __forceinline__ float lrelu(float x) { return x > 0.f ? x : NEG_SLOPE * x; }

// float atomic-max via int/uint ordering trick (init must be -inf)
__device__ __forceinline__ void atomicMaxF(float* addr, float val) {
  if (val >= 0.f) atomicMax((int*)addr, __float_as_int(val));
  else            atomicMin((unsigned int*)addr, (unsigned int)__float_as_int(val));
}

__device__ __forceinline__ void edge_sd(const int* __restrict__ ei, int e, int& s, int& d) {
  if (e < Ee) { s = ei[e]; d = ei[Ee + e]; } else { s = d = e - Ee; }
}

// ---------------- casts / transposes ----------------
__global__ void k_cast_bf16(const float* __restrict__ src, unsigned short* __restrict__ dst, int n) {
  int i = blockIdx.x * blockDim.x + threadIdx.x;
  if (i < n) dst[i] = f2bf(src[i]);
}

__global__ void k_transpose_bf16(const float* __restrict__ W, unsigned short* __restrict__ Wt,
                                 int rows, int cols) {
  int i = blockIdx.x * blockDim.x + threadIdx.x;
  if (i < rows * cols) { int r = i / cols, c = i % cols; Wt[(size_t)c * rows + r] = f2bf(W[i]); }
}

// ---------------- GEMM1: h1pre[8192,64] = Xbf[8192,512] @ W1 (W1t is bf16 [64,512]) ----------------
__global__ void k_gemm1(const unsigned short* __restrict__ Xu, const unsigned short* __restrict__ W1tu,
                        float* __restrict__ Hp) {
  const __bf16* Xb  = (const __bf16*)Xu;
  const __bf16* W1t = (const __bf16*)W1tu;
  int lane = threadIdx.x & 31;
  int wave = threadIdx.x >> 5;          // 4 waves -> 4 column tiles of 16
  int rowtile = blockIdx.x;             // 0..511
  int row = lane & 15, hi = lane >> 4;
  int col0 = wave * 16;
  const __bf16* pa = Xb  + (size_t)(rowtile * 16 + row) * FIN + hi * 16;
  const __bf16* pb = W1t + (size_t)(col0 + row) * FIN + hi * 16;
  v8f acc = {};
  for (int k0 = 0; k0 < FIN; k0 += 32) {
    v8bf a0 = *(const v8bf*)(pa + k0);        // K = k0 + hi*16 + 0..7
    v8bf a1 = *(const v8bf*)(pa + k0 + 8);    // K = k0 + hi*16 + 8..15
    v16bf b = *(const v16bf*)(pb + k0);       // K = k0 + hi*16 + j
    v16bf a;
    #pragma unroll
    for (int j = 0; j < 8; ++j) { a[j] = a0[j]; a[j + 8] = a1[j]; }
    acc = __builtin_amdgcn_wmma_f32_16x16x32_bf16(false, a, false, b, (short)0, acc, false, false);
  }
  #pragma unroll
  for (int r = 0; r < 8; ++r)
    Hp[(size_t)(rowtile * 16 + r + hi * 8) * HC1 + col0 + row] = acc[r];
}

// ---------------- GEMM2: zpre[8192,16] = H1bf[8192,64] @ W2 (W2t bf16 [16,64]) ----------------
__global__ void k_gemm2(const unsigned short* __restrict__ Hu, const unsigned short* __restrict__ W2tu,
                        float* __restrict__ Z) {
  const __bf16* Hb  = (const __bf16*)Hu;
  const __bf16* W2t = (const __bf16*)W2tu;
  int lane = threadIdx.x & 31;
  int wave = threadIdx.x >> 5;
  int rowtile = blockIdx.x * 4 + wave;  // 0..511
  int row = lane & 15, hi = lane >> 4;
  const __bf16* pa = Hb  + (size_t)(rowtile * 16 + row) * HC1 + hi * 16;
  const __bf16* pb = W2t + (size_t)row * HC1 + hi * 16;
  v8f acc = {};
  #pragma unroll
  for (int k0 = 0; k0 < HC1; k0 += 32) {
    v8bf a0 = *(const v8bf*)(pa + k0);
    v8bf a1 = *(const v8bf*)(pa + k0 + 8);
    v16bf b = *(const v16bf*)(pb + k0);
    v16bf a;
    #pragma unroll
    for (int j = 0; j < 8; ++j) { a[j] = a0[j]; a[j + 8] = a1[j]; }
    acc = __builtin_amdgcn_wmma_f32_16x16x32_bf16(false, a, false, b, (short)0, acc, false, false);
  }
  #pragma unroll
  for (int r = 0; r < 8; ++r)
    Z[(size_t)(rowtile * 16 + r + hi * 8) * NCn + row] = acc[r];
}

// ---------------- GEMM3: G[8192,8192] = z @ z^T, K=16 padded to 32 (Zbp bf16 [8192,32]) ----------------
__global__ void k_gemm3(const unsigned short* __restrict__ Zu, float* __restrict__ G) {
  const __bf16* Zb = (const __bf16*)Zu;
  int lane = threadIdx.x & 31;
  int wave = threadIdx.x >> 5;
  int tile = blockIdx.x * 8 + wave;     // 0 .. 512*512-1
  int tm = tile >> 9, tn = tile & 511;
  int row = lane & 15, hi = lane >> 4;
  // A: rows tm*16.., K padded; B: z^T columns tn*16.. -> contiguous rows of z
  const __bf16* pa = Zb + (size_t)(tm * 16 + row) * 32 + hi * 16;
  v8bf a0 = *(const v8bf*)pa;           // K = hi*16 + 0..7
  v8bf a1 = *(const v8bf*)(pa + 8);     // K = hi*16 + 8..15
  v16bf a;
  #pragma unroll
  for (int j = 0; j < 8; ++j) { a[j] = a0[j]; a[j + 8] = a1[j]; }
  v16bf b = *(const v16bf*)(Zb + (size_t)(tn * 16 + row) * 32 + hi * 16);  // K = hi*16 + j
  v8f acc = {};
  acc = __builtin_amdgcn_wmma_f32_16x16x32_bf16(false, a, false, b, (short)0, acc, false, false);
  #pragma unroll
  for (int r = 0; r < 8; ++r)
    G[(size_t)(tm * 16 + r + hi * 8) * Nn + tn * 16 + row] = acc[r];
}

// ---------------- attention coefficients ----------------
__global__ void k_att1(const float* __restrict__ Hp, const float* __restrict__ aS,
                       const float* __restrict__ aD, float* __restrict__ als, float* __restrict__ ald) {
  int i = blockIdx.x * blockDim.x + threadIdx.x;   // over N*H
  if (i >= Nn * Hh) return;
  int n = i >> 3, h = i & 7;
  const float* row = Hp + (size_t)n * HC1 + h * C1n;
  float s = 0.f, d = 0.f;
  #pragma unroll
  for (int c = 0; c < C1n; ++c) { s += row[c] * aS[h * C1n + c]; d += row[c] * aD[h * C1n + c]; }
  als[i] = s; ald[i] = d;
}

__global__ void k_att2(const float* __restrict__ Zp, const float* __restrict__ aS,
                       const float* __restrict__ aD, float* __restrict__ als, float* __restrict__ ald) {
  int n = blockIdx.x * blockDim.x + threadIdx.x;
  if (n >= Nn) return;
  const float* row = Zp + (size_t)n * NCn;
  float s = 0.f, d = 0.f;
  #pragma unroll
  for (int c = 0; c < NCn; ++c) { s += row[c] * aS[c]; d += row[c] * aD[c]; }
  als[n] = s; ald[n] = d;
}

// ---------------- init ----------------
__global__ void k_init1(float* __restrict__ m1, float* __restrict__ s1, float* __restrict__ out1) {
  int i = blockIdx.x * blockDim.x + threadIdx.x;
  if (i >= Nn * HC1) return;
  out1[i] = 0.f;
  if (i < Nn * Hh) { m1[i] = -INFINITY; s1[i] = 0.f; }
}
__global__ void k_init2(float* __restrict__ m2, float* __restrict__ s2, float* __restrict__ out2) {
  int i = blockIdx.x * blockDim.x + threadIdx.x;
  if (i >= Nn * NCn) return;
  out2[i] = 0.f;
  if (i < Nn) { m2[i] = -INFINITY; s2[i] = 0.f; }
}

// ---------------- layer-1 segment softmax + aggregate ----------------
__global__ void k_edge_max1(const int* __restrict__ ei, const float* __restrict__ als,
                            const float* __restrict__ ald, float* __restrict__ m1) {
  int e = blockIdx.x * blockDim.x + threadIdx.x;
  if (e >= ETot) return;
  int s, d; edge_sd(ei, e, s, d);
  #pragma unroll
  for (int h = 0; h < Hh; ++h)
    atomicMaxF(&m1[d * Hh + h], lrelu(als[s * Hh + h] + ald[d * Hh + h]));
}
__global__ void k_edge_sum1(const int* __restrict__ ei, const float* __restrict__ als,
                            const float* __restrict__ ald, const float* __restrict__ m1,
                            float* __restrict__ s1) {
  int e = blockIdx.x * blockDim.x + threadIdx.x;
  if (e >= ETot) return;
  int s, d; edge_sd(ei, e, s, d);
  #pragma unroll
  for (int h = 0; h < Hh; ++h) {
    float v = lrelu(als[s * Hh + h] + ald[d * Hh + h]);
    atomicAdd(&s1[d * Hh + h], expf(v - m1[d * Hh + h]));
  }
}
__global__ void k_edge_aggr1(const int* __restrict__ ei, const float* __restrict__ als,
                             const float* __restrict__ ald, const float* __restrict__ m1,
                             const float* __restrict__ s1, const float* __restrict__ Hp,
                             float* __restrict__ out1) {
  int e = blockIdx.x * blockDim.x + threadIdx.x;
  if (e >= ETot) return;
  int s, d; edge_sd(ei, e, s, d);
  #pragma unroll
  for (int h = 0; h < Hh; ++h) {
    float v = lrelu(als[s * Hh + h] + ald[d * Hh + h]);
    float alpha = expf(v - m1[d * Hh + h]) / s1[d * Hh + h];
    #pragma unroll
    for (int c = 0; c < C1n; ++c)
      atomicAdd(&out1[(size_t)d * HC1 + h * C1n + c], alpha * Hp[(size_t)s * HC1 + h * C1n + c]);
  }
}

// ---------------- bias + ELU + cast to bf16 ----------------
__global__ void k_elu_cast(const float* __restrict__ out1, const float* __restrict__ b1,
                           unsigned short* __restrict__ h1bf) {
  int i = blockIdx.x * blockDim.x + threadIdx.x;
  if (i >= Nn * HC1) return;
  float v = out1[i] + b1[i & (HC1 - 1)];
  float e = v > 0.f ? v : (expf(v) - 1.f);
  h1bf[i] = f2bf(e);
}

// ---------------- layer-2 segment softmax + aggregate ----------------
__global__ void k_edge_max2(const int* __restrict__ ei, const float* __restrict__ als,
                            const float* __restrict__ ald, float* __restrict__ m2) {
  int e = blockIdx.x * blockDim.x + threadIdx.x;
  if (e >= ETot) return;
  int s, d; edge_sd(ei, e, s, d);
  atomicMaxF(&m2[d], lrelu(als[s] + ald[d]));
}
__global__ void k_edge_sum2(const int* __restrict__ ei, const float* __restrict__ als,
                            const float* __restrict__ ald, const float* __restrict__ m2,
                            float* __restrict__ s2) {
  int e = blockIdx.x * blockDim.x + threadIdx.x;
  if (e >= ETot) return;
  int s, d; edge_sd(ei, e, s, d);
  atomicAdd(&s2[d], expf(lrelu(als[s] + ald[d]) - m2[d]));
}
__global__ void k_edge_aggr2(const int* __restrict__ ei, const float* __restrict__ als,
                             const float* __restrict__ ald, const float* __restrict__ m2,
                             const float* __restrict__ s2, const float* __restrict__ Zp,
                             float* __restrict__ out2) {
  int e = blockIdx.x * blockDim.x + threadIdx.x;
  if (e >= ETot) return;
  int s, d; edge_sd(ei, e, s, d);
  float alpha = expf(lrelu(als[s] + ald[d]) - m2[d]) / s2[d];
  #pragma unroll
  for (int c = 0; c < NCn; ++c)
    atomicAdd(&out2[(size_t)d * NCn + c], alpha * Zp[(size_t)s * NCn + c]);
}

// ---------------- bias + log_softmax to d_out, and zero-padded bf16 z for GEMM3 ----------------
__global__ void k_final(const float* __restrict__ out2, const float* __restrict__ b2,
                        float* __restrict__ lsm, unsigned short* __restrict__ zbfp) {
  int n = blockIdx.x * blockDim.x + threadIdx.x;
  if (n >= Nn) return;
  float zv[NCn]; float mx = -INFINITY;
  #pragma unroll
  for (int c = 0; c < NCn; ++c) { zv[c] = out2[(size_t)n * NCn + c] + b2[c]; mx = fmaxf(mx, zv[c]); }
  float se = 0.f;
  #pragma unroll
  for (int c = 0; c < NCn; ++c) se += expf(zv[c] - mx);
  float lse = logf(se) + mx;
  #pragma unroll
  for (int c = 0; c < NCn; ++c) lsm[(size_t)n * NCn + c] = zv[c] - lse;
  #pragma unroll
  for (int c = 0; c < NCn; ++c) zbfp[(size_t)n * 32 + c] = f2bf(zv[c]);
  #pragma unroll
  for (int c = NCn; c < 32; ++c) zbfp[(size_t)n * 32 + c] = 0;
}

extern "C" void kernel_launch(void* const* d_in, const int* in_sizes, int n_in,
                              void* d_out, int out_size, void* d_ws, size_t ws_size,
                              hipStream_t stream) {
  (void)in_sizes; (void)n_in; (void)out_size; (void)ws_size;
  const float* x   = (const float*)d_in[0];
  const int*   ei  = (const int*)  d_in[1];
  const float* W1  = (const float*)d_in[2];
  const float* aS1 = (const float*)d_in[3];
  const float* aD1 = (const float*)d_in[4];
  const float* b1  = (const float*)d_in[5];
  const float* W2  = (const float*)d_in[6];
  const float* aS2 = (const float*)d_in[7];
  const float* aD2 = (const float*)d_in[8];
  const float* b2  = (const float*)d_in[9];

  float* lsm = (float*)d_out;                    // [N, NC] log_softmax
  float* G   = lsm + (size_t)Nn * NCn;           // [N, N]  z @ z^T

  // carve workspace (256B aligned chunks)
  char* p = (char*)d_ws;
  auto carve = [&](size_t bytes) -> void* {
    void* r = (void*)p; p += (bytes + 255) & ~(size_t)255; return r;
  };
  unsigned short* xbf  = (unsigned short*)carve((size_t)Nn * FIN * 2);
  unsigned short* w1t  = (unsigned short*)carve((size_t)HC1 * FIN * 2);
  unsigned short* w2t  = (unsigned short*)carve((size_t)NCn * HC1 * 2);
  float* h1pre = (float*)carve((size_t)Nn * HC1 * 4);
  float* als1  = (float*)carve((size_t)Nn * Hh * 4);
  float* ald1  = (float*)carve((size_t)Nn * Hh * 4);
  float* m1    = (float*)carve((size_t)Nn * Hh * 4);
  float* s1    = (float*)carve((size_t)Nn * Hh * 4);
  float* out1  = (float*)carve((size_t)Nn * HC1 * 4);
  unsigned short* h1bf = (unsigned short*)carve((size_t)Nn * HC1 * 2);
  float* zpre  = (float*)carve((size_t)Nn * NCn * 4);
  float* als2  = (float*)carve((size_t)Nn * 4);
  float* ald2  = (float*)carve((size_t)Nn * 4);
  float* m2    = (float*)carve((size_t)Nn * 4);
  float* s2    = (float*)carve((size_t)Nn * 4);
  float* out2  = (float*)carve((size_t)Nn * NCn * 4);
  unsigned short* zbfp = (unsigned short*)carve((size_t)Nn * 32 * 2);

  const int B = 256;
  // casts / transposes
  k_cast_bf16<<<(Nn * FIN + B - 1) / B, B, 0, stream>>>(x, xbf, Nn * FIN);
  k_transpose_bf16<<<(FIN * HC1 + B - 1) / B, B, 0, stream>>>(W1, w1t, FIN, HC1);
  k_transpose_bf16<<<(HC1 * NCn + B - 1) / B, B, 0, stream>>>(W2, w2t, HC1, NCn);

  // layer 1
  k_gemm1<<<Nn / 16, 128, 0, stream>>>(xbf, w1t, h1pre);
  k_att1<<<(Nn * Hh) / B, B, 0, stream>>>(h1pre, aS1, aD1, als1, ald1);
  k_init1<<<(Nn * HC1) / B, B, 0, stream>>>(m1, s1, out1);
  int eg = (ETot + B - 1) / B;
  k_edge_max1 <<<eg, B, 0, stream>>>(ei, als1, ald1, m1);
  k_edge_sum1 <<<eg, B, 0, stream>>>(ei, als1, ald1, m1, s1);
  k_edge_aggr1<<<eg, B, 0, stream>>>(ei, als1, ald1, m1, s1, h1pre, out1);
  k_elu_cast<<<(Nn * HC1) / B, B, 0, stream>>>(out1, b1, h1bf);

  // layer 2
  k_gemm2<<<Nn / 64, 128, 0, stream>>>(h1bf, w2t, zpre);
  k_att2<<<Nn / B, B, 0, stream>>>(zpre, aS2, aD2, als2, ald2);
  k_init2<<<(Nn * NCn) / B, B, 0, stream>>>(m2, s2, out2);
  k_edge_max2 <<<eg, B, 0, stream>>>(ei, als2, ald2, m2);
  k_edge_sum2 <<<eg, B, 0, stream>>>(ei, als2, ald2, m2, s2);
  k_edge_aggr2<<<eg, B, 0, stream>>>(ei, als2, ald2, m2, s2, zpre, out2);

  // outputs
  k_final<<<Nn / B, B, 0, stream>>>(out2, b2, lsm, zbfp);
  k_gemm3<<<(512 * 512) / 8, 256, 0, stream>>>(zbfp, G);
}
